// NetEncoder0_55800215110121
// MI455X (gfx1250) — matrix operand order repaired and checked by reading, not compiled, not run on hardware
//
#include <hip/hip_runtime.h>
#include <hip/hip_bf16.h>
#include <math.h>

// ---------------- problem constants (from reference) ----------------
#define ND_C 4096
#define NE_C 8192
#define N_C  (ND_C + NE_C)   // 12288
#define D_C  128
#define L_C  16
#define H_C  8

// ---------------- WMMA types ----------------
typedef __attribute__((ext_vector_type(16))) __bf16 v16bf;
typedef __attribute__((ext_vector_type(8)))  float  v8f;

union BF16Frag {
    v16bf v;
    unsigned short u[16];
    uint4 q[2];
};

__device__ __forceinline__ unsigned short f2bf(float f) {
    unsigned int x = __float_as_uint(f);
    unsigned int lsb = (x >> 16) & 1u;
    x += 0x7FFFu + lsb;               // round-to-nearest-even
    return (unsigned short)(x >> 16);
}

// ---------------- weight convert + transpose: W(K x 128) fp32 -> WT(128 x K) bf16 ----------------
__global__ void cvt_transpose_kernel(const float* __restrict__ W,
                                     unsigned short* __restrict__ WT, int K) {
    int idx = blockIdx.x * blockDim.x + threadIdx.x;
    if (idx >= K * D_C) return;
    int k = idx >> 7;        // / 128
    int n = idx & 127;
    WT[(size_t)n * K + k] = f2bf(W[idx]);
}

// ---------------- fused WMMA GEMM (template-specialized: no runtime branches in hot loop) ----
// out0[row,col] = sum_k A[row,k]*W0[k,col] (+b0[col]) (*rowscale[row])
// DUAL: second half with W1/b1 -> out1, reusing the A fragment.
// Block: 256 threads = 8 waves. Block covers a 16-row M tile. Wave w covers
// output columns [16w, 16w+16) of each 128-column half. K must be %32, M %16.
template <bool DUAL, bool BIAS, bool SCALE>
__global__ __launch_bounds__(256)
void wmma_gemm_kernel(const float* __restrict__ A, int K,
                      const unsigned short* __restrict__ W0T,
                      const float* __restrict__ b0, float* __restrict__ out0,
                      const unsigned short* __restrict__ W1T,
                      const float* __restrict__ b1, float* __restrict__ out1,
                      const float* __restrict__ rowscale) {
    const int lane   = threadIdx.x & 31;
    const int wave   = threadIdx.x >> 5;       // 0..7
    const int m_base = blockIdx.x * 16;
    const int g      = lane >> 4;              // lane group 0/1
    const int ml     = lane & 15;              // A: row within tile
    const int nl     = lane & 15;              // B/C: col within tile
    const int n_base = wave * 16;

    v8f acc0, acc1;
    #pragma unroll
    for (int i = 0; i < 8; ++i) { acc0[i] = 0.f; acc1[i] = 0.f; }

    const float*          arow = A   + (size_t)(m_base + ml) * K;
    const unsigned short* w0p  = W0T + (size_t)(n_base + nl) * K;
    const unsigned short* w1p  = DUAL ? (W1T + (size_t)(n_base + nl) * K) : nullptr;

    for (int k0 = 0; k0 < K; k0 += 32) {
        // --- A fragment: 16x32 bf16, contiguous k order (B matches) ---
        // elem j : K = k0 + 16g + j
        const float* ap = arow + k0 + 16 * g;
        float4 x0 = *(const float4*)(ap);
        float4 x1 = *(const float4*)(ap + 4);
        float4 x2 = *(const float4*)(ap + 8);
        float4 x3 = *(const float4*)(ap + 12);
        BF16Frag af;
        af.v[0]  = (__bf16)x0.x; af.v[1]  = (__bf16)x0.y;
        af.v[2]  = (__bf16)x0.z; af.v[3]  = (__bf16)x0.w;
        af.v[4]  = (__bf16)x1.x; af.v[5]  = (__bf16)x1.y;
        af.v[6]  = (__bf16)x1.z; af.v[7]  = (__bf16)x1.w;
        af.v[8]  = (__bf16)x2.x; af.v[9]  = (__bf16)x2.y;
        af.v[10] = (__bf16)x2.z; af.v[11] = (__bf16)x2.w;
        af.v[12] = (__bf16)x3.x; af.v[13] = (__bf16)x3.y;
        af.v[14] = (__bf16)x3.z; af.v[15] = (__bf16)x3.w;

        // --- B fragment: 32x16 bf16; elem j: K = k0 + 16g + j, N = lane&15 ---
        // WT is [n][k] contiguous in k -> one 32B load per lane.
        BF16Frag bf0;
        const uint4* bp = (const uint4*)(w0p + k0 + 16 * g);
        bf0.q[0] = bp[0]; bf0.q[1] = bp[1];

        acc0 = __builtin_amdgcn_wmma_f32_16x16x32_bf16(
            false, af.v, false, bf0.v, (short)0, acc0, false, false);

        if (DUAL) {
            BF16Frag bf1;
            const uint4* bq = (const uint4*)(w1p + k0 + 16 * g);
            bf1.q[0] = bq[0]; bf1.q[1] = bq[1];
            acc1 = __builtin_amdgcn_wmma_f32_16x16x32_bf16(
                false, af.v, false, bf1.v, (short)0, acc1, false, false);
        }
    }

    // --- epilogue: C layout M = r + 8g, N = lane&15 ---
    const int col = n_base + nl;
    const float bias0 = BIAS ? b0[col] : 0.f;
    const float bias1 = (BIAS && DUAL) ? b1[col] : 0.f;
    #pragma unroll
    for (int r = 0; r < 8; ++r) {
        int row = m_base + r + 8 * g;
        float sc = SCALE ? rowscale[row] : 1.f;
        out0[(size_t)row * D_C + col] = (acc0[r] + bias0) * sc;
        if (DUAL)
            out1[(size_t)row * D_C + col] = (acc1[r] + bias1) * sc;
    }
}

// ---------------- attention scores: sd[node] = mean_h softmax_l(h0 . w_h)[0] ----------------
__global__ __launch_bounds__(128)
void att_score_kernel(const float* __restrict__ emb1, const int* __restrict__ node_seq,
                      const float* __restrict__ att_w_drug, const float* __restrict__ att_w_entity,
                      float* __restrict__ sd) {
    __shared__ float s_lds[H_C][L_C];
    __shared__ float p_lds[H_C];
    const int node = blockIdx.x;
    const int t = threadIdx.x;           // 0..127
    const int h = t >> 4, l = t & 15;
    const int seq = node_seq[node * L_C + l];
    const float* row = emb1 + (size_t)seq * D_C;
    const float* w = ((node < ND_C) ? att_w_drug : att_w_entity) + h * D_C;
    float s = 0.f;
    #pragma unroll 8
    for (int d = 0; d < D_C; ++d) s += row[d] * w[d];
    s_lds[h][l] = s;
    __syncthreads();
    if (l == 0) {
        float mx = -3.4e38f;
        for (int i = 0; i < L_C; ++i) mx = fmaxf(mx, s_lds[h][i]);
        float sum = 0.f, e0 = 0.f;
        for (int i = 0; i < L_C; ++i) {
            float e = __expf(s_lds[h][i] - mx);
            sum += e;
            if (i == 0) e0 = e;
        }
        p_lds[h] = e0 / sum;
    }
    __syncthreads();
    if (t == 0) {
        float m = 0.f;
        for (int i = 0; i < H_C; ++i) m += p_lds[i];
        sd[node] = m * (1.f / H_C);
    }
}

// ---------------- exclusion mask via exact stable-argsort rank ----------------
__global__ void rank_mask_kernel(const float* __restrict__ s, int count, int K,
                                 int* __restrict__ mask) {
    int i = blockIdx.x * blockDim.x + threadIdx.x;
    if (i >= count) return;
    float si = s[i];
    int rank = 0;
    for (int j = 0; j < count; ++j) {
        float sj = s[j];
        rank += (sj < si) || (sj == si && j < i);
    }
    mask[i] = (rank < K) ? 1 : 0;
}

// ---------------- overwrite excluded rows of feat with bias ----------------
__global__ __launch_bounds__(128)
void fixup_kernel(float* __restrict__ feat, const int* __restrict__ mask,
                  const float* __restrict__ b_d, const float* __restrict__ b_e) {
    int i = blockIdx.x, t = threadIdx.x;
    if (mask[i]) feat[(size_t)i * D_C + t] = (i < ND_C ? b_d : b_e)[t];
}

// ---------------- misc elementwise ----------------
__global__ void fillf_kernel(float* p, float v, int n) {
    int i = blockIdx.x * blockDim.x + threadIdx.x;
    if (i < n) p[i] = v;
}

__global__ void deg_kernel(const int* __restrict__ src, const int* __restrict__ dst,
                           float* __restrict__ degout, float* __restrict__ degin, int E) {
    int e = blockIdx.x * blockDim.x + threadIdx.x;
    if (e >= E) return;
    atomicAdd(&degout[src[e]], 1.f);
    atomicAdd(&degin[dst[e]], 1.f);
}

__global__ void rsqrt_kernel(float* d, int n) {
    int i = blockIdx.x * blockDim.x + threadIdx.x;
    if (i < n) {
        float v = d[i];
        v = v < 1.f ? 1.f : v;
        d[i] = rsqrtf(v);
    }
}

// ---------------- scatter: agg[dst] += x[src] over E edges + N self-loops ----------------
__global__ __launch_bounds__(128)
void scatter_kernel(const float* __restrict__ x, const int* __restrict__ src,
                    const int* __restrict__ dst, float* __restrict__ agg, int E) {
    int e = blockIdx.x, t = threadIdx.x;
    int s, d;
    if (e < E) { s = src[e]; d = dst[e]; }
    else       { s = d = e - E; }
    atomicAdd(&agg[(size_t)d * D_C + t], x[(size_t)s * D_C + t]);
}

// ---------------- gh = relu(agg * n_dst + b), in place ----------------
__global__ __launch_bounds__(128)
void gh_kernel(float* __restrict__ agg, const float* __restrict__ ndst,
               const float* __restrict__ b2) {
    int i = blockIdx.x, t = threadIdx.x;
    float v = agg[(size_t)i * D_C + t] * ndst[i] + b2[t];
    agg[(size_t)i * D_C + t] = v > 0.f ? v : 0.f;
}

// ---------------- final gather: out[row,:] = gh[node_seq0[row],:] ----------------
__global__ __launch_bounds__(128)
void gather_out_kernel(const float* __restrict__ gh, const int* __restrict__ seq0,
                       float* __restrict__ out) {
    int b = blockIdx.x, t = threadIdx.x;
    out[(size_t)b * D_C + t] = gh[(size_t)seq0[b] * D_C + t];
}

// ---------------- host side ----------------
static inline char* carve(char*& p, size_t bytes) {
    char* r = p;
    p += (bytes + 255) & ~(size_t)255;
    return r;
}

extern "C" void kernel_launch(void* const* d_in, const int* in_sizes, int n_in,
                              void* d_out, int out_size, void* d_ws, size_t ws_size,
                              hipStream_t stream) {
    const float* drug_feat    = (const float*)d_in[0];
    const float* entity_feat  = (const float*)d_in[1];
    const float* W_drug       = (const float*)d_in[2];
    const float* b_drug       = (const float*)d_in[3];
    const float* W_entity     = (const float*)d_in[4];
    const float* b_entity     = (const float*)d_in[5];
    const float* att_w_drug   = (const float*)d_in[6];
    const float* att_w_entity = (const float*)d_in[7];
    const float* W_drug1      = (const float*)d_in[8];
    const float* b_drug1      = (const float*)d_in[9];
    const float* W_entity1    = (const float*)d_in[10];
    const float* b_entity1    = (const float*)d_in[11];
    const float* gcn_W        = (const float*)d_in[12];
    const float* gcn_b        = (const float*)d_in[13];
    const int*   edge_index   = (const int*)d_in[14];
    const int*   node_seq     = (const int*)d_in[15];
    const int*   node_seq0    = (const int*)d_in[17];

    const int E = in_sizes[14] / 2;
    const int Kd = (int)(0.1 * ND_C);   // 409
    const int Ke = (int)(0.1 * NE_C);   // 819
    const float* gcn_W2 = gcn_W + 2 * D_C * D_C;   // only layer N_LAYERS-1 is used
    const float* gcn_b2 = gcn_b + 2 * D_C;

    // workspace carve (~31.5 MB)
    char* p = (char*)d_ws;
    float*          emb1   = (float*)carve(p, (size_t)N_C * D_C * 4);
    float*          feat   = (float*)carve(p, (size_t)N_C * D_C * 4);
    float*          x      = (float*)carve(p, (size_t)N_C * D_C * 4);
    float*          agg    = (float*)carve(p, (size_t)N_C * D_C * 4);
    unsigned short* WdT    = (unsigned short*)carve(p, (size_t)ND_C * D_C * 2);
    unsigned short* WeT    = (unsigned short*)carve(p, (size_t)NE_C * D_C * 2);
    unsigned short* Wd1T   = (unsigned short*)carve(p, (size_t)ND_C * D_C * 2);
    unsigned short* We1T   = (unsigned short*)carve(p, (size_t)NE_C * D_C * 2);
    unsigned short* WgT    = (unsigned short*)carve(p, (size_t)D_C * D_C * 2);
    float*          sd     = (float*)carve(p, (size_t)N_C * 4);
    int*            mask   = (int*)carve(p, (size_t)N_C * 4);
    float*          nsrc   = (float*)carve(p, (size_t)N_C * 4);
    float*          ndst   = (float*)carve(p, (size_t)N_C * 4);
    (void)ws_size; (void)n_in; (void)out_size;

    // 1) weights -> bf16, transposed to [n][k]
    cvt_transpose_kernel<<<(ND_C * D_C + 255) / 256, 256, 0, stream>>>(W_drug,   WdT,  ND_C);
    cvt_transpose_kernel<<<(NE_C * D_C + 255) / 256, 256, 0, stream>>>(W_entity, WeT,  NE_C);
    cvt_transpose_kernel<<<(ND_C * D_C + 255) / 256, 256, 0, stream>>>(W_drug1,  Wd1T, ND_C);
    cvt_transpose_kernel<<<(NE_C * D_C + 255) / 256, 256, 0, stream>>>(W_entity1, We1T, NE_C);
    cvt_transpose_kernel<<<(D_C * D_C + 255) / 256, 256, 0, stream>>>(gcn_W2,    WgT,  D_C);

    // 2) fused dual GEMMs: one pass over the big feature matrices produces both
    //    emb1 (= X@W + b) and feat-pre (= X@W1 + b1)
    wmma_gemm_kernel<true, true, false><<<ND_C / 16, 256, 0, stream>>>(
        drug_feat, ND_C, WdT, b_drug, emb1, Wd1T, b_drug1, feat, nullptr);
    wmma_gemm_kernel<true, true, false><<<NE_C / 16, 256, 0, stream>>>(
        entity_feat, NE_C, WeT, b_entity, emb1 + (size_t)ND_C * D_C,
        We1T, b_entity1, feat + (size_t)ND_C * D_C, nullptr);

    // 3) attention scores -> sd[N]
    att_score_kernel<<<N_C, 128, 0, stream>>>(emb1, node_seq, att_w_drug, att_w_entity, sd);

    // 4) exact stable-argsort-rank exclusion masks
    rank_mask_kernel<<<(ND_C + 255) / 256, 256, 0, stream>>>(sd,        ND_C, Kd, mask);
    rank_mask_kernel<<<(NE_C + 255) / 256, 256, 0, stream>>>(sd + ND_C, NE_C, Ke, mask + ND_C);

    // 5) excluded rows of feat collapse to their bias (zeroed input rows)
    fixup_kernel<<<N_C, 128, 0, stream>>>(feat, mask, b_drug1, b_entity1);

    // 6) degrees (init 1.0 accounts for self-loops), then ^-0.5 in place
    fillf_kernel<<<(N_C + 255) / 256, 256, 0, stream>>>(nsrc, 1.f, N_C);
    fillf_kernel<<<(N_C + 255) / 256, 256, 0, stream>>>(ndst, 1.f, N_C);
    deg_kernel<<<(E + 255) / 256, 256, 0, stream>>>(edge_index, edge_index + E, nsrc, ndst, E);
    rsqrt_kernel<<<(N_C + 255) / 256, 256, 0, stream>>>(nsrc, N_C);
    rsqrt_kernel<<<(N_C + 255) / 256, 256, 0, stream>>>(ndst, N_C);

    // 7) x = (feat @ gcn_W[2]) * n_src  (single-half WMMA GEMM with row scale)
    wmma_gemm_kernel<false, false, true><<<N_C / 16, 256, 0, stream>>>(
        feat, D_C, WgT, nullptr, x, nullptr, nullptr, nullptr, nsrc);

    // 8) agg = segment_sum(x[src], dst) over E edges + N self-loops
    fillf_kernel<<<(N_C * D_C + 255) / 256, 256, 0, stream>>>(agg, 0.f, N_C * D_C);
    scatter_kernel<<<E + N_C, 128, 0, stream>>>(x, edge_index, edge_index + E, agg, E);

    // 9) gh = relu(agg * n_dst + b2), in place
    gh_kernel<<<N_C, 128, 0, stream>>>(agg, ndst, gcn_b2);

    // 10) out = gh[node_seq0]  (drug block then entity block, contiguous)
    gather_out_kernel<<<N_C * L_C, 128, 0, stream>>>(agg, node_seq0, (float*)d_out);
}
